// RandomGraphAttention_5282809774390
// MI455X (gfx1250) — matrix operand order repaired and checked by reading, not compiled, not run on hardware
//
#include <hip/hip_runtime.h>
#include <hip/hip_bf16.h>

// ---------------- problem constants (fixed by reference) ----------------
#define BATCH   2
#define T_SEQ   4096
#define C_DIM   512
#define H_NUM   8
#define DK      64          // head dim; scale = 1/8 folded into Q

typedef __attribute__((ext_vector_type(16))) __bf16 v16bf;
typedef __attribute__((ext_vector_type(8)))  float  v8f;
typedef __attribute__((ext_vector_type(4)))  unsigned int tdm_v4u;
typedef __attribute__((ext_vector_type(8)))  int          tdm_v8i;
typedef __attribute__((ext_vector_type(4)))  int          tdm_v4i;

union FragU { uint4 u[2]; v16bf v; };

static __device__ __forceinline__ unsigned short f2bf(float f) {
  unsigned int u = __float_as_uint(f);
  unsigned int r = (u + 0x7FFFu + ((u >> 16) & 1u)) >> 16;   // RNE
  return (unsigned short)r;
}

// A-fragment, 16-bit 16x32 layout (ISA 7.12.2):
// lane group g = lane>>4; elements 0..7 hold K = 8g+0..7, elements 8..15 hold K = 16+8g+0..7
static __device__ __forceinline__ v16bf load_a_frag(const unsigned short* row, int g) {
  FragU f;
  f.u[0] = *(const uint4*)(row + 8 * g);
  f.u[1] = *(const uint4*)(row + 16 + 8 * g);
  return f.v;
}
// B-fragment, 16-bit 32x16 layout: lane holds column N = lane&15, elements 0..7 K = 8g+0..7, 8..15 K = 16+8g+0..7
static __device__ __forceinline__ v16bf load_b_frag(const unsigned short* col, int g) {
  FragU f;
  f.u[0] = *(const uint4*)(col + 8 * g);
  f.u[1] = *(const uint4*)(col + 16 + 8 * g);
  return f.v;
}

static __device__ __forceinline__ v8f wmma_bf16(v16bf a, v16bf b, v8f c) {
  return __builtin_amdgcn_wmma_f32_16x16x32_bf16(false, a, false, b, (short)0, c, false, false);
}

// row reductions across the 16 lanes that hold one C-frag row
static __device__ __forceinline__ float rmax16(float x) {
  x = fmaxf(x, __shfl_xor(x, 1, 16));
  x = fmaxf(x, __shfl_xor(x, 2, 16));
  x = fmaxf(x, __shfl_xor(x, 4, 16));
  x = fmaxf(x, __shfl_xor(x, 8, 16));
  return x;
}
static __device__ __forceinline__ float rsum16(float x) {
  x += __shfl_xor(x, 1, 16);
  x += __shfl_xor(x, 2, 16);
  x += __shfl_xor(x, 4, 16);
  x += __shfl_xor(x, 8, 16);
  return x;
}

// ---------------- Tensor Data Mover: 2-D tile global->LDS ----------------
// D# per ISA ch.8: group0 = {count/flags, lds_addr, global_addr, type=2},
// group1 = {wg_mask/data_size, tensor dims, tile dims, dim0 stride}. 2-D tile:
// tile_dim2=0, remaining groups zero. data_size code 1 = 2 bytes (bf16).
// This toolchain's builtin is the 6-arg form:
//   (uint32x4 g0, int32x8 g1, int32x4, int32x4, int32x8, i32 cpol)
#if __has_builtin(__builtin_amdgcn_tensor_load_to_lds)
#define HAVE_TDM 1
static __device__ __forceinline__ void tdm_load_2d(unsigned lds_off,
                                                   const unsigned short* gptr,
                                                   unsigned tile_d0, unsigned tile_d1,
                                                   unsigned long long stride_e) {
  unsigned long long ga = (unsigned long long)(size_t)gptr;
  tdm_v4u g0;
  g0.x = 1u;                                                  // count=1, user mode
  g0.y = lds_off;                                             // LDS byte address
  g0.z = (unsigned)ga;                                        // global_addr[31:0]
  g0.w = (unsigned)((ga >> 32) & 0x1FFFFFFull) | 0x80000000u; // addr[56:32] | type=2
  tdm_v8i g1;
  g1[0] = (int)(1u << 16);                                    // wg_mask=0, data_size=2B
  g1[1] = (int)((tile_d0 & 0xFFFFu) << 16);                   // tensor_dim0[15:0] (== tile_d0)
  g1[2] = (int)(((tile_d0 >> 16) & 0xFFFFu) | ((tile_d1 & 0xFFFFu) << 16)); // td0 hi | td1 lo
  g1[3] = (int)(((tile_d1 >> 16) & 0xFFFFu) | ((tile_d0 & 0xFFFFu) << 16)); // td1 hi | tile_dim0
  g1[4] = (int)(tile_d1 & 0xFFFFu);                           // tile_dim1 | tile_dim2=0
  g1[5] = (int)(unsigned)(stride_e & 0xFFFFFFFFull);          // tensor_dim0_stride[31:0]
  g1[6] = (int)(unsigned)((stride_e >> 32) & 0xFFFFull);      // stride[47:32] | td1_stride lo=0
  g1[7] = 0;
  tdm_v4i gz4 = {0, 0, 0, 0};
  tdm_v8i gz8 = {0, 0, 0, 0, 0, 0, 0, 0};
  __builtin_amdgcn_tensor_load_to_lds(g0, g1, gz4, gz4, gz8, 0);
}
#else
#define HAVE_TDM 0
#endif

// ---------------- prep kernels ----------------
__global__ void k_cvt_bf16(const float* __restrict__ src, unsigned short* __restrict__ dst, int n) {
  int i = blockIdx.x * blockDim.x + threadIdx.x;
  if (i < n) dst[i] = f2bf(src[i]);
}

// Wt[n][k] = bf16(W[k][n]) : makes B-fragments contiguous
__global__ void k_pack_w(const float* __restrict__ W, unsigned short* __restrict__ Wt) {
  int n = blockIdx.x * blockDim.x + threadIdx.x;   // contiguous read along n
  int k = blockIdx.y;
  Wt[n * C_DIM + k] = f2bf(W[k * C_DIM + n]);
}

// ---------------- fused QKV projection ----------------
// grid (BT/16=512, 8, 3)  block 128 (4 waves, each one 16-col tile)
__global__ __launch_bounds__(128) void k_qkv(const unsigned short* __restrict__ xb,
                                             const unsigned short* __restrict__ Wt,
                                             const float* __restrict__ bq,
                                             const float* __restrict__ bk,
                                             const float* __restrict__ bv,
                                             unsigned short* __restrict__ Qb,
                                             unsigned short* __restrict__ Kb,
                                             unsigned short* __restrict__ Vt) {
  const int wave = threadIdx.x >> 5, lane = threadIdx.x & 31;
  const int g = lane >> 4, nlo = lane & 15;
  const int rowtile = blockIdx.x;
  const int coltile = blockIdx.y * 4 + wave;
  const int mat = blockIdx.z;

  const unsigned short* W = Wt + (size_t)mat * C_DIM * C_DIM;
  const float* bias = (mat == 0) ? bq : (mat == 1) ? bk : bv;

  const unsigned short* arow = xb + (size_t)(rowtile * 16 + nlo) * C_DIM;
  const unsigned short* brow = W  + (size_t)(coltile * 16 + nlo) * C_DIM;

  v8f acc = {0.f, 0.f, 0.f, 0.f, 0.f, 0.f, 0.f, 0.f};
#pragma unroll
  for (int kb = 0; kb < C_DIM / 32; ++kb) {
    v16bf a = load_a_frag(arow + kb * 32, g);
    v16bf b = load_b_frag(brow + kb * 32, g);
    acc = wmma_bf16(a, b, acc);
  }
  const float bval = bias[coltile * 16 + nlo];
  const int cidx = coltile * 16 + nlo;
  const int h = cidx >> 6, d = cidx & 63;

#pragma unroll
  for (int r = 0; r < 8; ++r) {
    float v = acc[r] + bval;
    int grow = rowtile * 16 + r + 8 * g;
    int b_ = grow >> 12, t = grow & (T_SEQ - 1);
    size_t rc = ((size_t)(b_ * H_NUM + h) * T_SEQ + t) * DK + d;          // [B][H][T][dk]
    if (mat == 0)      Qb[rc] = f2bf(v * 0.125f);                         // fold 1/sqrt(dk)
    else if (mat == 1) Kb[rc] = f2bf(v);
    else               Vt[((size_t)(b_ * H_NUM + h) * DK + d) * T_SEQ + t] = f2bf(v); // [B][H][dk][T]
  }
}

// ---------------- attention (flash-style, multiplicative mask) ----------------
// grid (T/16/8=32, H, B)  block 256 (8 waves, each one 16-row query tile).
// K/V 32-key tiles staged in LDS once per block via TDM, double-buffered.
#define P_STRIDE 40   // ushorts per LDS P-row (pad for banks, keeps 16B align)
__global__ __launch_bounds__(256) void k_attn(const unsigned short* __restrict__ Qb,
                                              const unsigned short* __restrict__ Kb,
                                              const unsigned short* __restrict__ Vt,
                                              const float* __restrict__ mask,
                                              unsigned short* __restrict__ Ob) {
  __shared__ __align__(32) unsigned short kbuf[2][32 * DK];   // 32 keys x 64   (4KB each)
  __shared__ __align__(32) unsigned short vbuf[2][DK * 32];   // 64 d  x 32 key (4KB each)
  __shared__ __align__(32) unsigned short pbuf[8][16 * P_STRIDE];

  const int wave = threadIdx.x >> 5, lane = threadIdx.x & 31;
  const int g = lane >> 4, nlo = lane & 15;
  const int qtile = blockIdx.x * 8 + wave;
  const int h = blockIdx.y, b = blockIdx.z;
  const int q0 = qtile * 16;

  const size_t headoff = (size_t)(b * H_NUM + h) * T_SEQ * DK;
  const unsigned short* Qh = Qb + headoff;
  const unsigned short* Kh = Kb + headoff;
  const unsigned short* Vh = Vt + headoff;          // [dk][T]

  const unsigned short* qrow = Qh + (size_t)(q0 + nlo) * DK;
  const v16bf aq0 = load_a_frag(qrow, g);
  const v16bf aq1 = load_a_frag(qrow + 32, g);

  float m_run[8], l_run[8], alpha[8];
#pragma unroll
  for (int r = 0; r < 8; ++r) { m_run[r] = -3.0e38f; l_run[r] = 0.f; }
  v8f oacc[4];
#pragma unroll
  for (int j = 0; j < 4; ++j) oacc[j] = (v8f){0.f, 0.f, 0.f, 0.f, 0.f, 0.f, 0.f, 0.f};

  unsigned short* pl = pbuf[wave];
  const float* mbase = mask + (size_t)(q0 + 8 * g) * T_SEQ + nlo;

#if HAVE_TDM
  if (wave == 0) {            // wave-uniform branch: TDM ignores EXEC, issue once
    tdm_load_2d((unsigned)(size_t)&kbuf[0][0], Kh, DK, 32, DK);
    tdm_load_2d((unsigned)(size_t)&vbuf[0][0], Vh, 32, DK, T_SEQ);
    __builtin_amdgcn_s_wait_tensorcnt(0);
  }
  __syncthreads();
#else
  {
    const int t = threadIdx.x;                     // cooperative fallback staging
    *(uint4*)&kbuf[0][t * 8] = *(const uint4*)(Kh + t * 8);
    *(uint4*)&vbuf[0][t * 8] = *(const uint4*)(Vh + (size_t)(t >> 2) * T_SEQ + (t & 3) * 8);
    __syncthreads();
  }
#endif

  for (int kt = 0; kt < T_SEQ / 32; ++kt) {
    const int k0 = kt * 32;
    const int cur = kt & 1;

#if HAVE_TDM
    if (wave == 0 && kt + 1 < T_SEQ / 32) {        // prefetch next tile into other buffer
      const int kn = (kt + 1) * 32;
      tdm_load_2d((unsigned)(size_t)&kbuf[cur ^ 1][0], Kh + (size_t)kn * DK, DK, 32, DK);
      tdm_load_2d((unsigned)(size_t)&vbuf[cur ^ 1][0], Vh + kn, 32, DK, T_SEQ);
    }
#else
    if (kt + 1 < T_SEQ / 32) {
      const int kn = (kt + 1) * 32;
      const int t = threadIdx.x;
      *(uint4*)&kbuf[cur ^ 1][t * 8] = *(const uint4*)(Kh + (size_t)kn * DK + t * 8);
      *(uint4*)&vbuf[cur ^ 1][t * 8] =
          *(const uint4*)(Vh + (size_t)(t >> 2) * T_SEQ + kn + (t & 3) * 8);
    }
#endif

    const unsigned short* kb = kbuf[cur];
    const unsigned short* vb = vbuf[cur];

    const v8f z = {0.f, 0.f, 0.f, 0.f, 0.f, 0.f, 0.f, 0.f};
    v8f s0 = wmma_bf16(aq0, load_b_frag(kb + nlo * DK, g), z);
    s0 = wmma_bf16(aq1, load_b_frag(kb + nlo * DK + 32, g), s0);
    v8f s1 = wmma_bf16(aq0, load_b_frag(kb + (16 + nlo) * DK, g), z);
    s1 = wmma_bf16(aq1, load_b_frag(kb + (16 + nlo) * DK + 32, g), s1);

    // multiplicative mask BEFORE softmax (masked scores become 0, still in softmax)
    const float* mp = mbase + k0;
#pragma unroll
    for (int r = 0; r < 8; ++r) {
      s0[r] *= mp[(size_t)r * T_SEQ];
      s1[r] *= mp[(size_t)r * T_SEQ + 16];
    }

    // online softmax over all 32 columns
#pragma unroll
    for (int r = 0; r < 8; ++r) {
      float tm = rmax16(fmaxf(s0[r], s1[r]));
      float mnew = fmaxf(m_run[r], tm);
      float a_ = __expf(m_run[r] - mnew);
      float p0 = __expf(s0[r] - mnew);
      float p1 = __expf(s1[r] - mnew);
      l_run[r] = l_run[r] * a_ + rsum16(p0 + p1);
      m_run[r] = mnew;
      alpha[r] = a_;
      s0[r] = p0; s1[r] = p1;
    }
#pragma unroll
    for (int j = 0; j < 4; ++j)
#pragma unroll
      for (int r = 0; r < 8; ++r) oacc[j][r] *= alpha[r];

    // C-frag -> A-frag layout conversion through LDS (bf16 P tile, 16x32)
#pragma unroll
    for (int r = 0; r < 8; ++r) {
      int row = r + 8 * g;
      pl[row * P_STRIDE + nlo] = f2bf(s0[r]);
      pl[row * P_STRIDE + 16 + nlo] = f2bf(s1[r]);
    }
    v16bf ap = load_a_frag(pl + nlo * P_STRIDE, g);

    // P(16x32) x V(32x64): 4 WMMAs, V B-frags straight from LDS tile
#pragma unroll
    for (int j = 0; j < 4; ++j) {
      oacc[j] = wmma_bf16(ap, load_b_frag(vb + (j * 16 + nlo) * 32, g), oacc[j]);
    }

#if HAVE_TDM
    if (wave == 0) __builtin_amdgcn_s_wait_tensorcnt(0);  // DMA landed before barrier
#endif
    __syncthreads();
  }

  // normalize and store attention output as bf16, layout [B*T][C]
#pragma unroll
  for (int r = 0; r < 8; ++r) {
    const float inv = 1.0f / l_run[r];
    const int t = q0 + r + 8 * g;
    unsigned short* orow = Ob + (size_t)(b * T_SEQ + t) * C_DIM + h * DK;
#pragma unroll
    for (int j = 0; j < 4; ++j) orow[j * 16 + nlo] = f2bf(oacc[j][r] * inv);
  }
}

// ---------------- output projection: d_out = Ob @ Wo + bo (f32 out) ----------------
__global__ __launch_bounds__(128) void k_proj_out(const unsigned short* __restrict__ Ob,
                                                  const unsigned short* __restrict__ WtO,
                                                  const float* __restrict__ bo,
                                                  float* __restrict__ out) {
  const int wave = threadIdx.x >> 5, lane = threadIdx.x & 31;
  const int g = lane >> 4, nlo = lane & 15;
  const int rowtile = blockIdx.x;
  const int coltile = blockIdx.y * 4 + wave;

  const unsigned short* arow = Ob  + (size_t)(rowtile * 16 + nlo) * C_DIM;
  const unsigned short* brow = WtO + (size_t)(coltile * 16 + nlo) * C_DIM;

  v8f acc = {0.f, 0.f, 0.f, 0.f, 0.f, 0.f, 0.f, 0.f};
#pragma unroll
  for (int kb = 0; kb < C_DIM / 32; ++kb) {
    v16bf a = load_a_frag(arow + kb * 32, g);
    v16bf b = load_b_frag(brow + kb * 32, g);
    acc = wmma_bf16(a, b, acc);
  }
  const float bval = bo[coltile * 16 + nlo];
#pragma unroll
  for (int r = 0; r < 8; ++r) {
    int grow = rowtile * 16 + r + 8 * g;
    out[(size_t)grow * C_DIM + coltile * 16 + nlo] = acc[r] + bval;
  }
}

// ---------------- host launcher ----------------
extern "C" void kernel_launch(void* const* d_in, const int* in_sizes, int n_in,
                              void* d_out, int out_size, void* d_ws, size_t ws_size,
                              hipStream_t stream) {
  (void)in_sizes; (void)n_in; (void)out_size; (void)ws_size;
  const float* x    = (const float*)d_in[0];
  const float* mask = (const float*)d_in[1];
  const float* Wq   = (const float*)d_in[2];
  const float* bq   = (const float*)d_in[3];
  const float* Wk   = (const float*)d_in[4];
  const float* bk   = (const float*)d_in[5];
  const float* Wv   = (const float*)d_in[6];
  const float* bv   = (const float*)d_in[7];
  const float* Wo   = (const float*)d_in[8];
  const float* bo   = (const float*)d_in[9];
  float* out = (float*)d_out;

  const size_t BT = (size_t)BATCH * T_SEQ;           // 8192
  const size_t XN = BT * C_DIM;                      // 4,194,304
  const size_t HN = (size_t)BATCH * H_NUM * T_SEQ * DK;

  char* ws = (char*)d_ws;
  unsigned short* xb = (unsigned short*)(ws);                       size_t off = XN * 2;
  unsigned short* Wt = (unsigned short*)(ws + off);                 off += 4ull * C_DIM * C_DIM * 2;
  unsigned short* Qb = (unsigned short*)(ws + off);                 off += HN * 2;
  unsigned short* Kb = (unsigned short*)(ws + off);                 off += HN * 2;
  unsigned short* Vt = (unsigned short*)(ws + off);                 off += HN * 2;
  unsigned short* Ob = (unsigned short*)(ws + off);                 // XN*2 more (~42 MB total)

  // 1) x -> bf16
  k_cvt_bf16<<<dim3((unsigned)((XN + 255) / 256)), dim3(256), 0, stream>>>(x, xb, (int)XN);
  // 2) weights -> transposed bf16
  k_pack_w<<<dim3(C_DIM / 256, C_DIM), dim3(256), 0, stream>>>(Wq, Wt + 0ull * C_DIM * C_DIM);
  k_pack_w<<<dim3(C_DIM / 256, C_DIM), dim3(256), 0, stream>>>(Wk, Wt + 1ull * C_DIM * C_DIM);
  k_pack_w<<<dim3(C_DIM / 256, C_DIM), dim3(256), 0, stream>>>(Wv, Wt + 2ull * C_DIM * C_DIM);
  k_pack_w<<<dim3(C_DIM / 256, C_DIM), dim3(256), 0, stream>>>(Wo, Wt + 3ull * C_DIM * C_DIM);
  // 3) QKV projections (WMMA)
  k_qkv<<<dim3((unsigned)(BT / 16), C_DIM / 16 / 4, 3), dim3(128), 0, stream>>>(
      xb, Wt, bq, bk, bv, Qb, Kb, Vt);
  // 4) attention (WMMA + TDM-staged K/V + online softmax with multiplicative mask)
  k_attn<<<dim3(T_SEQ / 16 / 8, H_NUM, BATCH), dim3(256), 0, stream>>>(Qb, Kb, Vt, mask, Ob);
  // 5) output projection (WMMA, f32 out)
  k_proj_out<<<dim3((unsigned)(BT / 16), C_DIM / 16 / 4), dim3(128), 0, stream>>>(
      Ob, Wt + 3ull * C_DIM * C_DIM, bo, out);
}
